// PointNet2Regressor_16192026706633
// MI455X (gfx1250) — matrix operand order repaired and checked by reading, not compiled, not run on hardware
//
// PointNet++ (PointNet2Regressor) forward for MI455X / gfx1250, HIP + CDNA5 WMMA.
//
// Strategy (compile-only; reasoned from MI455X specs):
//  * All GEMM-shaped work runs through one fused kernel using V_WMMA_F32_16X16X32_F16
//    (f16 in, f32 accumulate). Activations in f16 (23.3 TB/s roofline: skinny-K layers
//    are ~40 FLOP/B, right at the f16 crossover).
//  * 64x64 block tile, 8 waves (4 M-strips x 2 N-halves); each wave reuses its A
//    fragment for two WMMAs.
//  * LDS tiles stored in fragment order (A row-major, B column-major) -> fragments
//    load as aligned ds_load_b128 pairs; 80 B row stride is bank-conflict-free.
//  * Staging is register double-buffered: global loads for slab i+1 are issued
//    before the WMMA phase of slab i, and use unpredicated fast paths when the
//    slab is fully in range (no per-element exec-mask serialization).
//  * Neighbor gather + centroid centering fused into the GEMM A stage; conv-bias +
//    BN (g * x * 1/sqrt(1+eps) + b) + ReLU fused into the epilogue.
//  * FPS / ball-query shared between the two SA towers (identical per reference).
//
// Input flattening assumption (documented): setup_inputs() dict insertion order,
// lists in index order, leaf layer dicts in insertion order {W,g,b}:
//   d_in[0]=points, d_in[1..36]=sa (4x3x{W,g,b}), d_in[37..63]=sa2 (3x3x3),
//   d_in[64..90]=fp (layer counts 3,2,2,2, x3),
//   d_in[91..96]=conv1_w, conv1_b, bn1_g, bn1_b, conv2_w, conv2_b.
//
// Workspace requirement: ~0.75 GB.

#include <hip/hip_runtime.h>
#include <hip/hip_bf16.h>
#include <cmath>
#include <cstdint>

typedef _Float16 h16;
typedef __attribute__((ext_vector_type(16))) _Float16 v16h;
typedef __attribute__((ext_vector_type(8)))  _Float16 v8h;
typedef __attribute__((ext_vector_type(8)))  float    v8f;

#define BM 64
#define BN 64
#define BK 32
#define LSTR 40   // LDS row stride in halves (80 B): conflict-free 16-lane phases

// ---- staging helpers: fast unpredicated path when slab fully in range ----
__device__ __forceinline__ void stage_b_regs(const float* __restrict__ W, int K, int N,
                                             int gn, int kbase, h16* tmp)
{
  if (kbase + 8 <= K && gn < N) {
    const float* wp = W + (size_t)kbase * N + gn;
#pragma unroll
    for (int e = 0; e < 8; ++e) tmp[e] = (h16)wp[(size_t)e * N];
  } else {
#pragma unroll
    for (int e = 0; e < 8; ++e) {
      int gk = kbase + e;
      tmp[e] = (h16)((gk < K && gn < N) ? W[(size_t)gk * N + gn] : 0.0f);
    }
  }
}

__device__ __forceinline__ void stage_a_regs(const h16* __restrict__ A,
    const float* __restrict__ sxyz, const float* __restrict__ cxyz,
    const h16* __restrict__ sfeat,
    int K, int Nsrc, int np, int Cf,
    int gmRow, int gb, int gs, int gj, int kbase, h16* tmp)
{
  if (A) {
    if (kbase + 8 <= K) {
      const h16* ap = A + (size_t)gmRow * K + kbase;
#pragma unroll
      for (int e = 0; e < 8; ++e) tmp[e] = ap[e];
    } else {
#pragma unroll
      for (int e = 0; e < 8; ++e) {
        int gk = kbase + e;
        tmp[e] = (gk < K) ? A[(size_t)gmRow * K + gk] : (h16)0.0f;
      }
    }
  } else {
    if (kbase >= 3 && kbase + 8 <= K) {
      const h16* fp = sfeat + ((size_t)gb * Nsrc + gj) * Cf + (kbase - 3);
#pragma unroll
      for (int e = 0; e < 8; ++e) tmp[e] = fp[e];
    } else {
#pragma unroll
      for (int e = 0; e < 8; ++e) {
        int gk = kbase + e;
        h16 v = (h16)0.0f;
        if (gk < K) {
          if (gk < 3)
            v = (h16)(sxyz[((size_t)gb * Nsrc + gj) * 3 + gk] -
                      cxyz[((size_t)gb * np   + gs) * 3 + gk]);
          else
            v = sfeat[((size_t)gb * Nsrc + gj) * Cf + (gk - 3)];
        }
        tmp[e] = v;
      }
    }
  }
}

// ---------------------------------------------------------------------------
// Fused GEMM: out = epilogue(A @ W). A: MxK f16 (plain or gathered on the fly).
// W: KxN f32 (params). Epilogue: +conv_bias, BN(g,b, *inv), ReLU, store f16/f32.
// M must be a multiple of 64 (true for every call site in this network).
// ---------------------------------------------------------------------------
__global__ __launch_bounds__(256) void gemm_fused(
    const h16* __restrict__ A,            // plain-mode A (MxK row major) or null
    const float* __restrict__ W,          // KxN row major
    const float* __restrict__ cbias,      // [N] or null
    const float* __restrict__ bng,        // [N] or null
    const float* __restrict__ bnb,        // [N] or null
    h16*  __restrict__ outH,              // MxN f16 or null
    float* __restrict__ outF,             // MxN f32 or null
    int M, int N, int K, int doRelu, float inv,
    // gather mode (A == null): row r -> (b, s, k), j = nidx[r]
    // c<3: sxyz[b,j,c]-cxyz[b,s,c]; else sfeat[b,j,c-3]
    const int*  __restrict__ nidx,
    const float* __restrict__ sxyz, const float* __restrict__ cxyz,
    const h16*  __restrict__ sfeat,
    int Nsrc, int ns, int np, int Cf)
{
  // A tile stored row-major in natural k order: lane-half h reads kk 16h..16h+15 (16 contiguous halves).
  // B tile stored column-major: Bs[col][kk] (lane-half reads kk 0..15 / 16..31).
  __shared__ __align__(16) h16 As[BM][LSTR];
  __shared__ __align__(16) h16 Bs[BN][LSTR];

  const int tid  = threadIdx.x;
  const int lane = tid & 31;
  const int wave = tid >> 5;
  const int wm   = wave & 3;    // M strip (16 rows)
  const int wh   = wave >> 2;   // N half (32 cols = 2 WMMA tiles)
  const int m0   = blockIdx.x * BM;
  const int n0   = blockIdx.y * BN;

  // per-thread staging coordinates (one 8-half segment per tile per thread)
  const int amm  = tid >> 2;            // A row within tile 0..63
  const int aseg = (tid & 3) << 3;      // A K-segment base 0/8/16/24
  const int ag   = aseg >> 3;
  const int aslot = ((ag & 1) << 1) | (ag >> 1);
  const int gmRow = m0 + amm;
  int gb = 0, gs = 0, gj = 0;
  if (!A) { gb = gmRow / (np * ns); gs = (gmRow / ns) % np; gj = nidx[gmRow]; }

  const int bnn  = tid >> 2;            // B column within tile 0..63
  const int bseg = (tid & 3) << 3;      // B K-segment base
  const int gn   = n0 + bnn;

  v8f acc0 = {}, acc1 = {};
  h16 atmp[8], btmp[8];

  stage_b_regs(W, K, N, gn, bseg, btmp);
  stage_a_regs(A, sxyz, cxyz, sfeat, K, Nsrc, np, Cf, gmRow, gb, gs, gj, aseg, atmp);

  for (int k0 = 0; k0 < K; k0 += BK) {
    // ---- commit staged registers to LDS (one ds_store_b128 each) ----
    *(v8h*)&Bs[bnn][bseg] = *(const v8h*)btmp;
    *(v8h*)&As[amm][aseg] = *(const v8h*)atmp;
    __syncthreads();

    // ---- issue next-slab global loads; they overlap the WMMA phase below ----
    const int kn = k0 + BK;
    if (kn < K) {
      if (kn + BK < K && tid < BK) {    // prefetch two slabs ahead
        __builtin_prefetch(W + (size_t)(kn + BK + tid) * N + n0, 0, 1);
      }
      stage_b_regs(W, K, N, gn, kn + bseg, btmp);
      stage_a_regs(A, sxyz, cxyz, sfeat, K, Nsrc, np, Cf, gmRow, gb, gs, gj, kn + aseg, atmp);
    }

    // ---- fragments: aligned 16 B LDS loads (ds_load_b128 pairs) ----
    const int rowA = wm * 16 + (lane & 15);
    const h16* ap  = &As[rowA][(lane < 16) ? 0 : 16];
    v8h a0 = *(const v8h*)ap, a1 = *(const v8h*)(ap + 8);
    v16h af = __builtin_shufflevector(a0, a1, 0,1,2,3,4,5,6,7,8,9,10,11,12,13,14,15);

    const int kbB = (lane < 16) ? 0 : 16;
    const h16* bp0 = &Bs[wh * 32 + (lane & 15)][kbB];
    const h16* bp1 = &Bs[wh * 32 + 16 + (lane & 15)][kbB];
    v8h b00 = *(const v8h*)bp0, b01 = *(const v8h*)(bp0 + 8);
    v8h b10 = *(const v8h*)bp1, b11 = *(const v8h*)(bp1 + 8);
    v16h bf0 = __builtin_shufflevector(b00, b01, 0,1,2,3,4,5,6,7,8,9,10,11,12,13,14,15);
    v16h bf1 = __builtin_shufflevector(b10, b11, 0,1,2,3,4,5,6,7,8,9,10,11,12,13,14,15);

    acc0 = __builtin_amdgcn_wmma_f32_16x16x32_f16(false, af, false, bf0, (short)0, acc0, false, false);
    acc1 = __builtin_amdgcn_wmma_f32_16x16x32_f16(false, af, false, bf1, (short)0, acc1, false, false);
    __syncthreads();
  }

  // ---- epilogue: D layout VGPR r, lane l -> M = r + 8*(l>=16), N = l&15 ----
  const int col0 = n0 + wh * 32 + (lane & 15);
  const int col1 = col0 + 16;
#pragma unroll
  for (int tile = 0; tile < 2; ++tile) {
    const int col = tile ? col1 : col0;
    if (col >= N) continue;
    const float cbv = cbias ? cbias[col] : 0.0f;
    const float gv  = bng ? bng[col] : 0.0f;
    const float bbv = bnb ? bnb[col] : 0.0f;
#pragma unroll
    for (int r = 0; r < 8; ++r) {
      int row = m0 + wm * 16 + r + ((lane >> 4) << 3);
      float v = (tile ? acc1[r] : acc0[r]) + cbv;
      if (bng) v = gv * (v * inv) + bbv;
      if (doRelu) v = v > 0.0f ? v : 0.0f;
      if (outH) outH[(size_t)row * N + col] = (h16)v;
      if (outF) outF[(size_t)row * N + col] = v;
    }
  }
}

// ---------------------------------------------------------------------------
// Farthest point sampling: one block per cloud, one thread per point
// (N in {512,256,128,64}). Matches lax.scan semantics.
// ---------------------------------------------------------------------------
__global__ void fps_kernel(const float* __restrict__ xyz, int* __restrict__ out,
                           int N, int npoint)
{
  __shared__ float rv[512];
  __shared__ int   ri[512];
  const int b = blockIdx.x, tid = threadIdx.x;
  const float* p = xyz + ((size_t)b * N + tid) * 3;
  const float px = p[0], py = p[1], pz = p[2];
  float mind = 1e10f;
  int far = 0;
  for (int it = 0; it < npoint; ++it) {
    if (tid == 0) out[b * npoint + it] = far;
    const float* q = xyz + ((size_t)b * N + far) * 3;
    float dx = px - q[0], dy = py - q[1], dz = pz - q[2];
    float d = dx * dx + dy * dy + dz * dz;
    mind = fminf(mind, d);
    rv[tid] = mind; ri[tid] = tid;
    __syncthreads();
    for (int s = N >> 1; s > 0; s >>= 1) {
      if (tid < s && rv[tid + s] > rv[tid]) { rv[tid] = rv[tid + s]; ri[tid] = ri[tid + s]; }
      __syncthreads();
    }
    far = ri[0];
    __syncthreads();
  }
}

__global__ void gather_xyz_kernel(const float* __restrict__ xyz, const int* __restrict__ cidx,
                                  float* __restrict__ nxyz, int N, int np, int total)
{
  int t = blockIdx.x * blockDim.x + threadIdx.x;
  if (t >= total) return;
  int c = t % 3, s = (t / 3) % np, b = t / (3 * np);
  nxyz[t] = xyz[((size_t)b * N + cidx[b * np + s]) * 3 + c];
}

// Ball query: smallest `ns` indices within radius (== top_k of -scores),
// padded with the first hit (reference pads with idx[...,:1]).
__global__ void ball_query_kernel(const float* __restrict__ xyz, const float* __restrict__ nxyz,
                                  int* __restrict__ nidx, int N, int np, int ns,
                                  float r2, int total)
{
  int t = blockIdx.x * blockDim.x + threadIdx.x;
  if (t >= total) return;
  int b = t / np;
  const float* c = nxyz + (size_t)t * 3;
  float cx = c[0], cy = c[1], cz = c[2];
  int* o = nidx + (size_t)t * ns;
  int cnt = 0;
  for (int j = 0; j < N && cnt < ns; ++j) {
    const float* q = xyz + ((size_t)b * N + j) * 3;
    float dx = q[0] - cx, dy = q[1] - cy, dz = q[2] - cz;
    if (dx * dx + dy * dy + dz * dz < r2) o[cnt++] = j;
  }
  int f = (cnt > 0) ? o[0] : 0;
  for (; cnt < ns; ++cnt) o[cnt] = f;
}

// 3-NN + inverse-distance weights for feature propagation.
__global__ void knn3_kernel(const float* __restrict__ txyz, const float* __restrict__ sxyz,
                            int n1, int n2, int* __restrict__ idx3, float* __restrict__ w3,
                            int total)
{
  int t = blockIdx.x * blockDim.x + threadIdx.x;
  if (t >= total) return;
  int b = t / n1;
  const float* p = txyz + (size_t)t * 3;
  float px = p[0], py = p[1], pz = p[2];
  float d0 = 1e30f, d1 = 1e30f, d2 = 1e30f;
  int i0 = 0, i1 = 0, i2 = 0;
  for (int j = 0; j < n2; ++j) {
    const float* q = sxyz + ((size_t)b * n2 + j) * 3;
    float dx = px - q[0], dy = py - q[1], dz = pz - q[2];
    float d = dx * dx + dy * dy + dz * dz;
    if (d < d0)      { d2 = d1; i2 = i1; d1 = d0; i1 = i0; d0 = d; i0 = j; }
    else if (d < d1) { d2 = d1; i2 = i1; d1 = d;  i1 = j; }
    else if (d < d2) { d2 = d;  i2 = j; }
  }
  float r0 = 1.0f / (d0 + 1e-8f), r1 = 1.0f / (d1 + 1e-8f), r2v = 1.0f / (d2 + 1e-8f);
  float s = r0 + r1 + r2v;
  idx3[t * 3 + 0] = i0; idx3[t * 3 + 1] = i1; idx3[t * 3 + 2] = i2;
  w3[t * 3 + 0] = r0 / s; w3[t * 3 + 1] = r1 / s; w3[t * 3 + 2] = r2v / s;
}

// interp = sum_j w_j * f2[idx_j] ; out = concat(interp, f1)  (f16 rows)
__global__ void interp_concat_kernel(const h16* __restrict__ f2, int n2, int C2,
                                     const int* __restrict__ idx3, const float* __restrict__ w3,
                                     const h16* __restrict__ f1, int C1,
                                     h16* __restrict__ out, int n1, long long total)
{
  long long t = (long long)blockIdx.x * blockDim.x + threadIdx.x;
  if (t >= total) return;
  const int C = C2 + C1;
  long long row = t / C;
  int c = (int)(t - row * C);
  if (c < C2) {
    int b = (int)(row / n1);
    const int*   id = idx3 + row * 3;
    const float* w  = w3 + row * 3;
    float v = w[0] * (float)f2[((size_t)b * n2 + id[0]) * C2 + c]
            + w[1] * (float)f2[((size_t)b * n2 + id[1]) * C2 + c]
            + w[2] * (float)f2[((size_t)b * n2 + id[2]) * C2 + c];
    out[t] = (h16)v;
  } else {
    out[t] = f1[(size_t)row * C1 + (c - C2)];
  }
}

__global__ void maxpool_kernel(const h16* __restrict__ in, h16* __restrict__ outH,
                               float* __restrict__ outF, int ns, int C, long long total)
{
  long long t = (long long)blockIdx.x * blockDim.x + threadIdx.x;
  if (t >= total) return;
  long long row = t / C;
  int c = (int)(t - row * C);
  const h16* p = in + (size_t)row * ns * C + c;
  float m = (float)p[0];
  for (int j = 1; j < ns; ++j) m = fmaxf(m, (float)p[(size_t)j * C]);
  if (outH) outH[t] = (h16)m;
  if (outF) outF[t] = m;
}

__global__ void softmax_mask_kernel(const float* __restrict__ logits, float* __restrict__ xout,
                                    h16* __restrict__ mask, int total)
{
  int t = blockIdx.x * blockDim.x + threadIdx.x;
  if (t >= total) return;
  float a = logits[2 * t], b = logits[2 * t + 1];
  float m = fmaxf(a, b);
  float ea = expf(a - m), eb = expf(b - m);
  float s = ea + eb;
  xout[2 * t] = ea / s;
  xout[2 * t + 1] = eb / s;
  mask[t] = (h16)((a < b) ? 1.0f : 0.0f);
}

__global__ void concat_mask_kernel(const h16* __restrict__ mask, const h16* __restrict__ f0,
                                   h16* __restrict__ out, int C0, long long total)
{
  long long t = (long long)blockIdx.x * blockDim.x + threadIdx.x;
  if (t >= total) return;
  const int C = C0 + 1;
  long long row = t / C;
  int c = (int)(t - row * C);
  out[t] = (c == 0) ? mask[row] : f0[(size_t)row * C0 + (c - 1)];
}

__global__ void concat_xyzfeat_kernel(const float* __restrict__ xyz, const h16* __restrict__ f,
                                      h16* __restrict__ out, int Cf, long long total)
{
  long long t = (long long)blockIdx.x * blockDim.x + threadIdx.x;
  if (t >= total) return;
  const int C = 3 + Cf;
  long long row = t / C;
  int c = (int)(t - row * C);
  out[t] = (c < 3) ? (h16)xyz[(size_t)row * 3 + c] : f[(size_t)row * Cf + (c - 3)];
}

// ---------------------------------------------------------------------------
extern "C" void kernel_launch(void* const* d_in, const int* in_sizes, int n_in,
                              void* d_out, int out_size, void* d_ws, size_t ws_size,
                              hipStream_t stream)
{
  (void)in_sizes; (void)n_in; (void)out_size; (void)ws_size;

  const int B = 128;               // B*T point clouds
  const float inv = 1.0f / sqrtf(1.0f + 1e-5f);

  // ---- parameter mapping (see file header for order assumption) ----
  const float* points = (const float*)d_in[0];   // (128, 512, 3) == xyz
  int pi = 1;
  const float *saW[4][3], *saG[4][3], *saBb[4][3];
  for (int m = 0; m < 4; ++m)
    for (int l = 0; l < 3; ++l) {
      saW[m][l]  = (const float*)d_in[pi++];
      saG[m][l]  = (const float*)d_in[pi++];
      saBb[m][l] = (const float*)d_in[pi++];
    }
  const float *s2W[3][3], *s2G[3][3], *s2Bb[3][3];
  for (int m = 0; m < 3; ++m)
    for (int l = 0; l < 3; ++l) {
      s2W[m][l]  = (const float*)d_in[pi++];
      s2G[m][l]  = (const float*)d_in[pi++];
      s2Bb[m][l] = (const float*)d_in[pi++];
    }
  const int fpCnt[4] = {3, 2, 2, 2};
  const float *fpW[4][3], *fpG[4][3], *fpBb[4][3];
  for (int m = 0; m < 4; ++m)
    for (int l = 0; l < fpCnt[m]; ++l) {
      fpW[m][l]  = (const float*)d_in[pi++];
      fpG[m][l]  = (const float*)d_in[pi++];
      fpBb[m][l] = (const float*)d_in[pi++];
    }
  const float* conv1_w = (const float*)d_in[pi++];
  const float* conv1_b = (const float*)d_in[pi++];
  const float* bn1_g   = (const float*)d_in[pi++];
  const float* bn1_b   = (const float*)d_in[pi++];
  const float* conv2_w = (const float*)d_in[pi++];
  const float* conv2_b = (const float*)d_in[pi++];

  // ---- workspace bump allocator ----
  char* base = (char*)d_ws;
  size_t off = 0;
  auto alloc = [&](size_t bytes) -> void* {
    off = (off + 255) & ~(size_t)255;
    void* p = base + off;
    off += bytes;
    return p;
  };

  int* cidx1 = (int*)alloc((size_t)B * 256 * 4);
  int* cidx2 = (int*)alloc((size_t)B * 128 * 4);
  int* cidx3 = (int*)alloc((size_t)B * 64 * 4);
  int* cidx4 = (int*)alloc((size_t)B * 16 * 4);
  int* nidx1 = (int*)alloc((size_t)B * 256 * 32 * 4);
  int* nidx2 = (int*)alloc((size_t)B * 128 * 32 * 4);
  int* nidx3 = (int*)alloc((size_t)B * 64 * 32 * 4);
  int* nidx4 = (int*)alloc((size_t)B * 16 * 32 * 4);
  float* xyz1 = (float*)alloc((size_t)B * 256 * 3 * 4);
  float* xyz2 = (float*)alloc((size_t)B * 128 * 3 * 4);
  float* xyz3 = (float*)alloc((size_t)B * 64 * 3 * 4);
  float* xyz4 = (float*)alloc((size_t)B * 16 * 3 * 4);
  h16* feat1 = (h16*)alloc((size_t)B * 256 * 128 * 2);
  h16* feat2 = (h16*)alloc((size_t)B * 128 * 256 * 2);
  h16* feat3 = (h16*)alloc((size_t)B * 64 * 512 * 2);
  h16* feat4 = (h16*)alloc((size_t)B * 16 * 1024 * 2);
  h16* nf3   = (h16*)alloc((size_t)B * 64 * 512 * 2);
  h16* nf2   = (h16*)alloc((size_t)B * 128 * 512 * 2);
  h16* nf1   = (h16*)alloc((size_t)B * 256 * 256 * 2);
  h16* f0    = (h16*)alloc((size_t)B * 512 * 256 * 2);
  h16* hfeat = (h16*)alloc((size_t)B * 512 * 257 * 2);
  h16* maskb = (h16*)alloc((size_t)B * 512 * 2);
  float* logits = (float*)alloc((size_t)B * 512 * 2 * 4);
  h16* hfeat1 = (h16*)alloc((size_t)B * 256 * 128 * 2);
  h16* hfeat2 = (h16*)alloc((size_t)B * 128 * 256 * 2);
  int*   kidx = (int*)alloc((size_t)B * 512 * 3 * 4);
  float* kw   = (float*)alloc((size_t)B * 512 * 3 * 4);
  h16* cA = (h16*)alloc((size_t)B * 128 * 259 * 2);
  h16* T0 = (h16*)alloc((size_t)268435456);
  h16* T1 = (h16*)alloc((size_t)268435456);

  auto blocks = [](long long total) { return dim3((unsigned)((total + 255) / 256)); };

  auto gemm = [&](const h16* A, const float* W, const float* cb, const float* g,
                  const float* bb, h16* oH, float* oF, int M, int N, int K, int relu,
                  const int* nid, const float* sx, const float* cx, const h16* sf,
                  int Nsrc, int ns, int np, int Cf) {
    dim3 grid(M / BM, (N + BN - 1) / BN);
    gemm_fused<<<grid, 256, 0, stream>>>(A, W, cb, g, bb, oH, oF, M, N, K, relu, inv,
                                         nid, sx, cx, sf, Nsrc, ns, np, Cf);
  };
  auto mlp = [&](const h16* A, int m, int l, const float* (*Wt)[3], const float* (*Gt)[3],
                 const float* (*Bt)[3], h16* oH, int M, int N, int K,
                 const int* nid, const float* sx, const float* cx, const h16* sf,
                 int Nsrc, int ns, int np, int Cf) {
    gemm(A, Wt[m][l], nullptr, Gt[m][l], Bt[m][l], oH, nullptr, M, N, K, 1,
         nid, sx, cx, sf, Nsrc, ns, np, Cf);
  };

  // ================= sampling (shared by both SA towers) =================
  fps_kernel<<<B, 512, 0, stream>>>(points, cidx1, 512, 256);
  gather_xyz_kernel<<<blocks(B*256*3), 256, 0, stream>>>(points, cidx1, xyz1, 512, 256, B*256*3);
  ball_query_kernel<<<blocks(B*256), 256, 0, stream>>>(points, xyz1, nidx1, 512, 256, 32, 0.04f, B*256);

  fps_kernel<<<B, 256, 0, stream>>>(xyz1, cidx2, 256, 128);
  gather_xyz_kernel<<<blocks(B*128*3), 256, 0, stream>>>(xyz1, cidx2, xyz2, 256, 128, B*128*3);
  ball_query_kernel<<<blocks(B*128), 256, 0, stream>>>(xyz1, xyz2, nidx2, 256, 128, 32, 0.16f, B*128);

  fps_kernel<<<B, 128, 0, stream>>>(xyz2, cidx3, 128, 64);
  gather_xyz_kernel<<<blocks(B*64*3), 256, 0, stream>>>(xyz2, cidx3, xyz3, 128, 64, B*64*3);
  ball_query_kernel<<<blocks(B*64), 256, 0, stream>>>(xyz2, xyz3, nidx3, 128, 64, 32, 0.16f, B*64);

  fps_kernel<<<B, 64, 0, stream>>>(xyz3, cidx4, 64, 16);
  gather_xyz_kernel<<<blocks(B*16*3), 256, 0, stream>>>(xyz3, cidx4, xyz4, 64, 16, B*16*3);
  ball_query_kernel<<<blocks(B*16), 256, 0, stream>>>(xyz3, xyz4, nidx4, 64, 16, 32, 0.64f, B*16);

  // ================= SA tower 1 =================
  { int M = B*256*32;   // SA1: gather(3) -> 32 -> 64 -> 128, maxpool over 32
    mlp(nullptr, 0,0, saW,saG,saBb, T0, M, 32, 3,  nidx1, points, xyz1, nullptr, 512,32,256,0);
    mlp(T0,      0,1, saW,saG,saBb, T1, M, 64, 32, nullptr,nullptr,nullptr,nullptr,0,0,0,0);
    mlp(T1,      0,2, saW,saG,saBb, T0, M, 128,64, nullptr,nullptr,nullptr,nullptr,0,0,0,0);
    maxpool_kernel<<<blocks((long long)B*256*128), 256, 0, stream>>>(T0, feat1, nullptr, 32, 128, (long long)B*256*128);
  }
  { int M = B*128*32;   // SA2: gather(131) -> 128 -> 128 -> 256
    mlp(nullptr, 1,0, saW,saG,saBb, T0, M, 128, 131, nidx2, xyz1, xyz2, feat1, 256,32,128,128);
    mlp(T0,      1,1, saW,saG,saBb, T1, M, 128, 128, nullptr,nullptr,nullptr,nullptr,0,0,0,0);
    mlp(T1,      1,2, saW,saG,saBb, T0, M, 256, 128, nullptr,nullptr,nullptr,nullptr,0,0,0,0);
    maxpool_kernel<<<blocks((long long)B*128*256), 256, 0, stream>>>(T0, feat2, nullptr, 32, 256, (long long)B*128*256);
  }
  { int M = B*64*32;    // SA3: gather(259) -> 256 -> 256 -> 512
    mlp(nullptr, 2,0, saW,saG,saBb, T0, M, 256, 259, nidx3, xyz2, xyz3, feat2, 128,32,64,256);
    mlp(T0,      2,1, saW,saG,saBb, T1, M, 256, 256, nullptr,nullptr,nullptr,nullptr,0,0,0,0);
    mlp(T1,      2,2, saW,saG,saBb, T0, M, 512, 256, nullptr,nullptr,nullptr,nullptr,0,0,0,0);
    maxpool_kernel<<<blocks((long long)B*64*512), 256, 0, stream>>>(T0, feat3, nullptr, 32, 512, (long long)B*64*512);
  }
  { int M = B*16*32;    // SA4: gather(515) -> 512 -> 512 -> 1024
    mlp(nullptr, 3,0, saW,saG,saBb, T0, M, 512, 515, nidx4, xyz3, xyz4, feat3, 64,32,16,512);
    mlp(T0,      3,1, saW,saG,saBb, T1, M, 512, 512, nullptr,nullptr,nullptr,nullptr,0,0,0,0);
    mlp(T1,      3,2, saW,saG,saBb, T0, M, 1024,512, nullptr,nullptr,nullptr,nullptr,0,0,0,0);
    maxpool_kernel<<<blocks((long long)B*16*1024), 256, 0, stream>>>(T0, feat4, nullptr, 32, 1024, (long long)B*16*1024);
  }

  // ================= FP tower =================
  knn3_kernel<<<blocks(B*64), 256, 0, stream>>>(xyz3, xyz4, 64, 16, kidx, kw, B*64);
  interp_concat_kernel<<<blocks((long long)B*64*1536), 256, 0, stream>>>(feat4, 16, 1024, kidx, kw, feat3, 512, T0, 64, (long long)B*64*1536);
  mlp(T0, 3,0, fpW,fpG,fpBb, T1, B*64, 512, 1536, nullptr,nullptr,nullptr,nullptr,0,0,0,0);
  mlp(T1, 3,1, fpW,fpG,fpBb, nf3, B*64, 512, 512, nullptr,nullptr,nullptr,nullptr,0,0,0,0);

  knn3_kernel<<<blocks(B*128), 256, 0, stream>>>(xyz2, xyz3, 128, 64, kidx, kw, B*128);
  interp_concat_kernel<<<blocks((long long)B*128*768), 256, 0, stream>>>(nf3, 64, 512, kidx, kw, feat2, 256, T0, 128, (long long)B*128*768);
  mlp(T0, 2,0, fpW,fpG,fpBb, T1, B*128, 512, 768, nullptr,nullptr,nullptr,nullptr,0,0,0,0);
  mlp(T1, 2,1, fpW,fpG,fpBb, nf2, B*128, 512, 512, nullptr,nullptr,nullptr,nullptr,0,0,0,0);

  knn3_kernel<<<blocks(B*256), 256, 0, stream>>>(xyz1, xyz2, 256, 128, kidx, kw, B*256);
  interp_concat_kernel<<<blocks((long long)B*256*640), 256, 0, stream>>>(nf2, 128, 512, kidx, kw, feat1, 128, T0, 256, (long long)B*256*640);
  mlp(T0, 1,0, fpW,fpG,fpBb, T1, B*256, 512, 640, nullptr,nullptr,nullptr,nullptr,0,0,0,0);
  mlp(T1, 1,1, fpW,fpG,fpBb, nf1, B*256, 256, 512, nullptr,nullptr,nullptr,nullptr,0,0,0,0);

  knn3_kernel<<<blocks(B*512), 256, 0, stream>>>(points, xyz1, 512, 256, kidx, kw, B*512);
  interp_concat_kernel<<<blocks((long long)B*512*256), 256, 0, stream>>>(nf1, 256, 256, kidx, kw, nullptr, 0, T0, 512, (long long)B*512*256);
  mlp(T0, 0,0, fpW,fpG,fpBb, T1, B*512, 256, 256, nullptr,nullptr,nullptr,nullptr,0,0,0,0);
  mlp(T1, 0,1, fpW,fpG,fpBb, T0, B*512, 256, 256, nullptr,nullptr,nullptr,nullptr,0,0,0,0);
  mlp(T0, 0,2, fpW,fpG,fpBb, f0, B*512, 256, 256, nullptr,nullptr,nullptr,nullptr,0,0,0,0);

  // ========= head: conv1(+bias)+BN+ReLU, conv2(+bias), softmax =========
  gemm(f0, conv1_w, conv1_b, bn1_g, bn1_b, T1, nullptr, B*512, 128, 256, 1,
       nullptr,nullptr,nullptr,nullptr,0,0,0,0);
  gemm(T1, conv2_w, conv2_b, nullptr, nullptr, nullptr, logits, B*512, 2, 128, 0,
       nullptr,nullptr,nullptr,nullptr,0,0,0,0);
  softmax_mask_kernel<<<blocks(B*512), 256, 0, stream>>>(logits, (float*)d_out, maskb, B*512);

  // ================= SA tower 2 =================
  concat_mask_kernel<<<blocks((long long)B*512*257), 256, 0, stream>>>(maskb, f0, hfeat, 256, (long long)B*512*257);
  { int M = B*256*32;   // sa2[0]: gather(260) -> 128 -> 128 -> 128 (reuses nidx1/xyz1)
    mlp(nullptr, 0,0, s2W,s2G,s2Bb, T0, M, 128, 260, nidx1, points, xyz1, hfeat, 512,32,256,257);
    mlp(T0,      0,1, s2W,s2G,s2Bb, T1, M, 128, 128, nullptr,nullptr,nullptr,nullptr,0,0,0,0);
    mlp(T1,      0,2, s2W,s2G,s2Bb, T0, M, 128, 128, nullptr,nullptr,nullptr,nullptr,0,0,0,0);
    maxpool_kernel<<<blocks((long long)B*256*128), 256, 0, stream>>>(T0, hfeat1, nullptr, 32, 128, (long long)B*256*128);
  }
  { int M = B*128*32;   // sa2[1]: gather(131) -> 128 -> 128 -> 256 (reuses nidx2/xyz2)
    mlp(nullptr, 1,0, s2W,s2G,s2Bb, T0, M, 128, 131, nidx2, xyz1, xyz2, hfeat1, 256,32,128,128);
    mlp(T0,      1,1, s2W,s2G,s2Bb, T1, M, 128, 128, nullptr,nullptr,nullptr,nullptr,0,0,0,0);
    mlp(T1,      1,2, s2W,s2G,s2Bb, T0, M, 256, 128, nullptr,nullptr,nullptr,nullptr,0,0,0,0);
    maxpool_kernel<<<blocks((long long)B*128*256), 256, 0, stream>>>(T0, hfeat2, nullptr, 32, 256, (long long)B*128*256);
  }
  // sa2[2]: group-all concat(xyz2, hfeat2)=259 -> 256 -> 512 -> 1024, maxpool(128) -> human
  concat_xyzfeat_kernel<<<blocks((long long)B*128*259), 256, 0, stream>>>(xyz2, hfeat2, cA, 256, (long long)B*128*259);
  mlp(cA, 2,0, s2W,s2G,s2Bb, T0, B*128, 256, 259, nullptr,nullptr,nullptr,nullptr,0,0,0,0);
  mlp(T0, 2,1, s2W,s2G,s2Bb, T1, B*128, 512, 256, nullptr,nullptr,nullptr,nullptr,0,0,0,0);
  mlp(T1, 2,2, s2W,s2G,s2Bb, T0, B*128, 1024, 512, nullptr,nullptr,nullptr,nullptr,0,0,0,0);
  maxpool_kernel<<<blocks((long long)B*1024), 256, 0, stream>>>(T0, nullptr, (float*)d_out + (size_t)B*512*2, 128, 1024, (long long)B*1024);
}
